// minGRU_11811160064387
// MI455X (gfx1250) — matrix operand order repaired and checked by reading, compiled, not run on hardware
//
#include <hip/hip_runtime.h>
#include <math.h>

typedef float  v8f   __attribute__((ext_vector_type(8)));
typedef __bf16 v16bf __attribute__((ext_vector_type(16)));

#define M_TOTAL   65536        // B*S = 8*8192
#define K_TOTAL   512          // D
#define N_TOTAL   1024         // 2*D
#define KC        32           // K window = one bf16 WMMA K
#define NWIN      (K_TOTAL / KC)
#define PITCH     40           // bf16 row pitch: 80B, 16B-aligned, bank-spread
#define SCHUNK    256          // scan chunk length
#define NCHUNK    32           // 8192 / 256
#define NCHAN     4096         // B*D

// ------------------------------------------------------- bf16x3 split -------
__device__ __forceinline__ unsigned fbits(float x) { return __builtin_bit_cast(unsigned, x); }
__device__ __forceinline__ float    bitsf(unsigned u) { return __builtin_bit_cast(float, u); }

// float4 -> packed bf16 hi pair-dwords + residual-lo pair-dwords (truncation
// split: hi exact in bf16, r = x - hi exact in fp32, lo = trunc-bf16(r))
__device__ __forceinline__ void split4(float4 v, uint2& hp, uint2& lp)
{
    unsigned h0 = fbits(v.x) & 0xFFFF0000u, h1 = fbits(v.y) & 0xFFFF0000u;
    unsigned h2 = fbits(v.z) & 0xFFFF0000u, h3 = fbits(v.w) & 0xFFFF0000u;
    float r0 = v.x - bitsf(h0), r1 = v.y - bitsf(h1);
    float r2 = v.z - bitsf(h2), r3 = v.w - bitsf(h3);
    hp.x = (h0 >> 16) | h1;
    hp.y = (h2 >> 16) | h3;
    lp.x = (fbits(r0) >> 16) | (fbits(r1) & 0xFFFF0000u);
    lp.y = (fbits(r2) >> 16) | (fbits(r3) & 0xFFFF0000u);
}

// ISA 16-bit A 16x32 layout: lane (M=lane&15) holds K = {8g..8g+7, 16+8g..16+8g+7}
__device__ __forceinline__ v16bf load_frag_a(const unsigned short* rowp, int g)
{
    struct S { uint4 a; uint4 b; } s;
    s.a = *reinterpret_cast<const uint4*>(rowp + 8 * g);
    s.b = *reinterpret_cast<const uint4*>(rowp + 16 + 8 * g);
    return __builtin_bit_cast(v16bf, s);
}
// ISA 16-bit B 32x16 layout: lane (N=lane&15) holds K = {8g..8g+7, 16+8g..16+8g+7}
__device__ __forceinline__ v16bf load_frag_b(const unsigned short* colp, int g)
{
    struct S { uint4 a; uint4 b; } s;
    s.a = *reinterpret_cast<const uint4*>(colp + 8 * g);
    s.b = *reinterpret_cast<const uint4*>(colp + 16 + 8 * g);
    return __builtin_bit_cast(v16bf, s);
}

// ---------------------------------------------------------------- GEMM ------
// HG[m, e] = sum_k X[m, k] * W[e, k]   (fp32 via bf16x3 on v_wmma_f32_16x16x32_bf16)
// Grid: x = N-blocks (16) fastest so co-resident blocks share one A tile in L2.
__global__ __launch_bounds__(256)
void gemm_hg_kernel(const float* __restrict__ X, const float* __restrict__ W,
                    float* __restrict__ HG)
{
    // double-buffered hi/lo planes: (2*128 + 2*64) * PITCH bf16 * 2 bufs = 60KB
    __shared__ unsigned short AsH[2][128 * PITCH];
    __shared__ unsigned short AsL[2][128 * PITCH];
    __shared__ unsigned short BsH[2][ 64 * PITCH];
    __shared__ unsigned short BsL[2][ 64 * PITCH];

    const int n0   = blockIdx.x * 64;      // fastest dim: N (16 blocks)
    const int m0   = blockIdx.y * 128;
    const int tid  = threadIdx.x;
    const int wave = tid >> 5;
    const int lane = tid & 31;
    const int lrow = lane & 15;
    const int lhi  = lane >> 4;
    const int mq   = wave >> 1;            // wave tile: 32M x 32N, 4x2 waves
    const int nq   = wave & 1;

    v8f acc[2][2] = {{v8f{}, v8f{}}, {v8f{}, v8f{}}};

    float4 aReg[4], bReg[2];

    // stage helpers (A: 128x32 floats / 256 thr = 4 float4; B: 64x32 = 2 float4)
    auto load_window = [&](int k0) {
        #pragma unroll
        for (int it = 0; it < 4; ++it) {
            int idx = (it * 256 + tid) * 4, row = idx >> 5, col = idx & 31;
            aReg[it] = *reinterpret_cast<const float4*>(
                &X[(size_t)(m0 + row) * K_TOTAL + k0 + col]);
        }
        #pragma unroll
        for (int it = 0; it < 2; ++it) {
            int idx = (it * 256 + tid) * 4, row = idx >> 5, col = idx & 31;
            bReg[it] = *reinterpret_cast<const float4*>(
                &W[(size_t)(n0 + row) * K_TOTAL + k0 + col]);
        }
    };
    auto store_window = [&](int buf) {
        #pragma unroll
        for (int it = 0; it < 4; ++it) {
            int idx = (it * 256 + tid) * 4, row = idx >> 5, col = idx & 31;
            uint2 hp, lp;
            split4(aReg[it], hp, lp);
            *reinterpret_cast<uint2*>(&AsH[buf][row * PITCH + col]) = hp;
            *reinterpret_cast<uint2*>(&AsL[buf][row * PITCH + col]) = lp;
        }
        #pragma unroll
        for (int it = 0; it < 2; ++it) {
            int idx = (it * 256 + tid) * 4, row = idx >> 5, col = idx & 31;
            uint2 hp, lp;
            split4(bReg[it], hp, lp);
            *reinterpret_cast<uint2*>(&BsH[buf][row * PITCH + col]) = hp;
            *reinterpret_cast<uint2*>(&BsL[buf][row * PITCH + col]) = lp;
        }
    };

    load_window(0);
    store_window(0);
    __syncthreads();

    int cur = 0;
    for (int w = 0; w < NWIN; ++w) {
        if (w + 1 < NWIN) load_window((w + 1) * KC);   // prefetch into registers

        // A fragments for the wave's two M sub-rows (hi and lo planes)
        v16bf ah[2], al[2];
        #pragma unroll
        for (int ms = 0; ms < 2; ++ms) {
            int arow = (mq * 32 + ms * 16 + lrow) * PITCH;
            ah[ms] = load_frag_a(&AsH[cur][arow], lhi);
            al[ms] = load_frag_a(&AsL[cur][arow], lhi);
        }
        #pragma unroll
        for (int ns = 0; ns < 2; ++ns) {
            int bcol = (nq * 32 + ns * 16 + lrow) * PITCH;
            v16bf bh = load_frag_b(&BsH[cur][bcol], lhi);
            v16bf bl = load_frag_b(&BsL[cur][bcol], lhi);
            #pragma unroll
            for (int ms = 0; ms < 2; ++ms) {
                // bf16x3: hi*hi + lo*hi + hi*lo, fp32 accumulate
                acc[ms][ns] = __builtin_amdgcn_wmma_f32_16x16x32_bf16(false, ah[ms], false, bh, (short)0, acc[ms][ns], false, false);
                acc[ms][ns] = __builtin_amdgcn_wmma_f32_16x16x32_bf16(false, al[ms], false, bh, (short)0, acc[ms][ns], false, false);
                acc[ms][ns] = __builtin_amdgcn_wmma_f32_16x16x32_bf16(false, ah[ms], false, bl, (short)0, acc[ms][ns], false, false);
            }
        }

        if (w + 1 < NWIN) {
            store_window(cur ^ 1);   // safe: nobody reads cur^1 this window
            __syncthreads();
            cur ^= 1;
        }
    }

    // Epilogue: C/D 16x16 f32 layout -> lane holds N = lane&15, M = v + 8*(lane>>4)
    // Non-temporal: HG is a 256MB stream; keep x/W resident in L2 instead.
    #pragma unroll
    for (int ms = 0; ms < 2; ++ms) {
        #pragma unroll
        for (int v = 0; v < 8; ++v) {
            int m = m0 + mq * 32 + ms * 16 + v + 8 * lhi;
            size_t base = (size_t)m * N_TOTAL + n0 + nq * 32 + lrow;
            __builtin_nontemporal_store(acc[ms][0][v], &HG[base +  0]);
            __builtin_nontemporal_store(acc[ms][1][v], &HG[base + 16]);
        }
    }
}

// ------------------------------------------------------------- scan math ----
__device__ __forceinline__ float softplus_f(float x) {
    return fmaxf(x, 0.0f) + log1pf(__expf(-fabsf(x)));
}
__device__ __forceinline__ float log_g_f(float x) {
    return (x >= 0.0f) ? __logf(x + 0.5f) : -softplus_f(-x);
}
__device__ __forceinline__ float logaddexp_f(float x, float y) {
    float m = fmaxf(x, y);
    float d = fminf(x, y) - m;
    return m + log1pf(__expf(d));   // exp(-inf)=0 handles one-sided -inf
}

// Phase 1: per (b, chunk, d) lane-task, reduce 256 steps to one (A, V) pair.
__global__ __launch_bounds__(256)
void scan_phase1(const float* __restrict__ HG,
                 float* __restrict__ sumA, float* __restrict__ sumV)
{
    int g     = blockIdx.x * 256 + threadIdx.x;   // 0..131071
    int d     = g & 511;
    int chunk = (g >> 9) & 31;
    int b     = g >> 14;
    const float* p = HG + ((size_t)b * 8192 + (size_t)chunk * SCHUNK) * N_TOTAL + d;
    float A = 0.0f, V = -__builtin_inff();
    for (int i = 0; i < SCHUNK; ++i) {
        float h  = __builtin_nontemporal_load(p);
        float gt = __builtin_nontemporal_load(p + 512);
        float lc = -softplus_f(gt);                    // log_coeffs
        float lv = -softplus_f(-gt) + log_g_f(h);      // log_values
        V = logaddexp_f(V + lc, lv);
        A += lc;
        p += N_TOTAL;
    }
    int ch = b * 512 + d;
    sumA[ch * NCHUNK + chunk] = A;
    sumV[ch * NCHUNK + chunk] = V;
}

// Phase 2: one wave32 per channel; lane = chunk. Inclusive shuffle-scan of the
// semiring operator, then shift to an exclusive prefix (V component only).
__global__ __launch_bounds__(256)
void scan_phase2(const float* __restrict__ sumA, const float* __restrict__ sumV,
                 float* __restrict__ prefV)
{
    int wave = threadIdx.x >> 5;
    int lane = threadIdx.x & 31;
    int ch   = blockIdx.x * 8 + wave;     // 0..4095
    float A = sumA[ch * NCHUNK + lane];
    float V = sumV[ch * NCHUNK + lane];
    #pragma unroll
    for (int off = 1; off < 32; off <<= 1) {
        float pA = __shfl_up(A, off, 32);
        float pV = __shfl_up(V, off, 32);
        if (lane >= off) {
            V = logaddexp_f(pV + A, V);
            A = pA + A;
        }
    }
    float eV = __shfl_up(V, 1, 32);
    if (lane == 0) eV = -__builtin_inff();
    prefV[ch * NCHUNK + lane] = eV;
}

// Phase 3: replay chunk with exclusive prefix, emit out = exp(log_h).
__global__ __launch_bounds__(256)
void scan_phase3(const float* __restrict__ HG, const float* __restrict__ prefV,
                 float* __restrict__ OUT, float* __restrict__ OUT2)
{
    int g     = blockIdx.x * 256 + threadIdx.x;
    int d     = g & 511;
    int chunk = (g >> 9) & 31;
    int b     = g >> 14;
    int ch    = b * 512 + d;
    float pV  = prefV[ch * NCHUNK + chunk];
    const float* p = HG + ((size_t)b * 8192 + (size_t)chunk * SCHUNK) * N_TOTAL + d;
    float* o = OUT + ((size_t)b * 8192 + (size_t)chunk * SCHUNK) * 512 + d;
    float A = 0.0f, V = -__builtin_inff(), last = 0.0f;
    for (int i = 0; i < SCHUNK; ++i) {
        float h  = __builtin_nontemporal_load(p);
        float gt = __builtin_nontemporal_load(p + 512);
        float lc = -softplus_f(gt);
        float lv = -softplus_f(-gt) + log_g_f(h);
        V = logaddexp_f(V + lc, lv);
        A += lc;
        last = __expf(logaddexp_f(pV + A, V));
        __builtin_nontemporal_store(last, o);
        p += N_TOTAL;
        o += 512;
    }
    if (chunk == NCHUNK - 1) OUT2[ch] = last;   // next_prev_hidden (s = 8191)
}

// ----------------------------------------------------------------- launch ---
extern "C" void kernel_launch(void* const* d_in, const int* in_sizes, int n_in,
                              void* d_out, int out_size, void* d_ws, size_t ws_size,
                              hipStream_t stream)
{
    const float* x = (const float*)d_in[0];   // (8, 8192, 512)
    const float* W = (const float*)d_in[1];   // (1024, 512)
    float* out  = (float*)d_out;                              // (8,8192,512)
    float* out2 = out + (size_t)8 * 8192 * 512;               // (8,1,512)

    float* hg    = (float*)d_ws;                              // 64M floats
    float* sumA  = hg   + (size_t)M_TOTAL * N_TOTAL;          // 4096*32
    float* sumV  = sumA + (size_t)NCHAN * NCHUNK;
    float* prefV = sumV + (size_t)NCHAN * NCHUNK;

    dim3 ggrid(N_TOTAL / 64, M_TOTAL / 128);                  // 16 x 512, N fastest
    gemm_hg_kernel<<<ggrid, dim3(256), 0, stream>>>(x, W, hg);
    scan_phase1 <<<dim3(512), dim3(256), 0, stream>>>(hg, sumA, sumV);
    scan_phase2 <<<dim3(512), dim3(256), 0, stream>>>(sumA, sumV, prefV);
    scan_phase3 <<<dim3(512), dim3(256), 0, stream>>>(hg, prefV, out, out2);
}
